// GodModeCausalGRU_6674379178373
// MI455X (gfx1250) — matrix operand order repaired and checked by reading, not compiled, not run on hardware
//
#include <hip/hip_runtime.h>

// ---------------- CDNA5 WMMA types / helpers -------------------------------
typedef __attribute__((ext_vector_type(16))) __bf16 v16bf;
typedef __attribute__((ext_vector_type(8)))  float  v8f;

__device__ __forceinline__ v8f wmma_bf16(v16bf a, v16bf b, v8f c) {
  // (neg_a, A, neg_b, B, c_mod, C, reuse_a, reuse_b)
  return __builtin_amdgcn_wmma_f32_16x16x32_bf16(false, a, false, b, (short)0, c,
                                                 false, false);
}

// CDNA5 async global->LDS copy (ASYNCcnt-tracked), 16B per lane.
// VDST = VGPR holding LDS byte offset, VADDR = 64-bit global address.
__device__ __forceinline__ void async_b128(unsigned lds_off, const void* gaddr) {
  asm volatile("global_load_async_to_lds_b128 %0, %1, off"
               :: "v"(lds_off), "v"((unsigned long long)(size_t)gaddr)
               : "memory");
}
__device__ __forceinline__ void wait_async0() {
  asm volatile("s_wait_asynccnt 0x0" ::: "memory");
}
// Generic LDS pointer -> 32-bit LDS offset (low 32 bits of shared aperture addr)
__device__ __forceinline__ unsigned lds_off(const void* p) {
  return (unsigned)(size_t)p;
}

// A fragment: 16x32 bf16 tile at (m0,k0) from row-major LDS [m][k] (stride elems).
// Contiguous k order used here (k = 16h + s): lanes 0-15 -> M=lane, K=0..15; lanes 16-31 -> M=lane-16, K=16..31.
__device__ __forceinline__ v16bf load_a_frag(const __bf16* As, int stride,
                                             int m0, int k0, int lane) {
  v16bf a;
  const __bf16* p = As + (long)(m0 + (lane & 15)) * stride + k0 + ((lane >> 4) << 4);
#pragma unroll
  for (int v = 0; v < 8; ++v) {
    int k = v << 1;
    a[2 * v]     = p[k];
    a[2 * v + 1] = p[k + 1];
  }
  return a;
}

// B fragment: B = W^T where W is row-major [n][k] in LDS. Lanes 0-15: N=lane,
// K=0..15; lanes 16-31: N=lane-16, K=16..31. 32B-aligned contiguous per lane.
__device__ __forceinline__ v16bf load_b_frag(const __bf16* Ws, int stride,
                                             int n0, int k0, int lane) {
  int n = n0 + (lane & 15);
  int k = k0 + ((lane >> 4) << 4);
  return *(const v16bf*)(Ws + (long)n * stride + k);
}

__device__ __forceinline__ float wave_sum(float v) {
#pragma unroll
  for (int off = 16; off; off >>= 1) v += __shfl_xor(v, off, 32);
  return v;
}

// ---------------- generic bf16 GEMM: C(f32) = A(bf16) * W^T + bias ---------
// Double-buffered LDS tiles filled with ASYNC global->LDS copies.
#define BM 64
#define BN 64
#define BK 32
#define SA 40   // padded LDS k-stride for A (80B rows keep 16B-aligned chunks)
#define SB 32

__global__ void gemm_bf16_kernel(const __bf16* __restrict__ A, long lda,
                                 const __bf16* __restrict__ W, int ldw,
                                 const float* __restrict__ bias,
                                 float* __restrict__ C, long ldc,
                                 long M, int N, int K) {
  __shared__ __bf16 As[2][BM * SA];
  __shared__ __bf16 Bs[2][BN * SB];
  const int tid = threadIdx.x, lane = tid & 31, wave = tid >> 5;
  const long m0 = (long)blockIdx.y * BM;
  const int  n0 = blockIdx.x * BN;
  const int  wm = (wave & 3) * 16;   // wave m-tile
  const int  wn = (wave >> 2) * 32;  // wave n-strip (two 16-wide tiles)
  v8f acc0 = {}; v8f acc1 = {};

  const int ar = tid >> 2;           // 0..63 staging row
  const int ac = (tid & 3) * 8;      // 0,8,16,24 staging col
  const long am = m0 + ar;
  const bool a_ok = (am < M);        // stale rows only feed unstored outputs

  auto stage = [&](int buf, int k0) {
    if (a_ok)
      async_b128(lds_off(&As[buf][ar * SA + ac]), A + am * lda + k0 + ac);
    async_b128(lds_off(&Bs[buf][ar * SB + ac]),
               W + (long)(n0 + ar) * ldw + k0 + ac);
  };

  const int nk = K / BK;
  stage(0, 0);
  for (int kc = 0; kc < nk; ++kc) {
    const int cur = kc & 1;
    wait_async0();
    __syncthreads();
    if (kc + 1 < nk) stage(cur ^ 1, (kc + 1) * BK);
    v16bf af  = load_a_frag(As[cur], SA, wm, 0, lane);
    v16bf bf0 = load_b_frag(Bs[cur], SB, wn, 0, lane);
    v16bf bf1 = load_b_frag(Bs[cur], SB, wn + 16, 0, lane);
    acc0 = wmma_bf16(af, bf0, acc0);
    acc1 = wmma_bf16(af, bf1, acc1);
    __syncthreads();
  }
  const int nl = lane & 15, mo = (lane >> 4) * 8;
#pragma unroll
  for (int p = 0; p < 8; ++p) {
    long m = m0 + wm + mo + p;
    if (m < M) {
      int c0 = n0 + wn + nl, c1 = c0 + 16;
      C[m * ldc + c0] = acc0[p] + (bias ? bias[c0] : 0.f);
      C[m * ldc + c1] = acc1[p] + (bias ? bias[c1] : 0.f);
    }
  }
}

// ---------------- causal conv1d (k=7) as 7 shifted accumulated GEMMs -------
// Weights pre-repacked to bf16 [w][o][i] so B-tiles are contiguous async copies.
__global__ void conv_gemm_kernel(const __bf16* __restrict__ feat, long lda,
                                 const __bf16* __restrict__ winw,  // [7][128][128]
                                 const float* __restrict__ win_b,
                                 float* __restrict__ C, long ldc,
                                 long M, int T) {
  __shared__ __bf16 As[2][BM * SA];
  __shared__ __bf16 Bs[2][BN * SB];
  const int tid = threadIdx.x, lane = tid & 31, wave = tid >> 5;
  const long m0 = (long)blockIdx.y * BM;
  const int  n0 = blockIdx.x * BN;
  const int  wm = (wave & 3) * 16;
  const int  wn = (wave >> 2) * 32;
  v8f acc0 = {}; v8f acc1 = {};
  const int ar = tid >> 2, ac = (tid & 3) * 8;
  const long am = m0 + ar;
  const int  at = (int)(am % (long)T);

  auto stage = [&](int buf, int it) {
    const int w = it >> 2, k0 = (it & 3) * BK;
    const int shift = 6 - w;                 // x_t convolves feat[t - shift]
    const bool ok = (am < M) && (at >= shift);
    if (ok) {
      async_b128(lds_off(&As[buf][ar * SA + ac]),
                 feat + (am - shift) * lda + k0 + ac);
    } else {
      *(uint4*)&As[buf][ar * SA + ac] = make_uint4(0u, 0u, 0u, 0u);
    }
    async_b128(lds_off(&Bs[buf][ar * SB + ac]),
               winw + (long)w * 128 * 128 + (long)(n0 + ar) * 128 + k0 + ac);
  };

  const int nIter = 7 * 4;  // taps x K-chunks
  stage(0, 0);
  for (int it = 0; it < nIter; ++it) {
    const int cur = it & 1;
    wait_async0();
    __syncthreads();
    if (it + 1 < nIter) stage(cur ^ 1, it + 1);
    v16bf af  = load_a_frag(As[cur], SA, wm, 0, lane);
    v16bf bf0 = load_b_frag(Bs[cur], SB, wn, 0, lane);
    v16bf bf1 = load_b_frag(Bs[cur], SB, wn + 16, 0, lane);
    acc0 = wmma_bf16(af, bf0, acc0);
    acc1 = wmma_bf16(af, bf1, acc1);
    __syncthreads();
  }
  const int nl = lane & 15, mo = (lane >> 4) * 8;
#pragma unroll
  for (int p = 0; p < 8; ++p) {
    long m = m0 + wm + mo + p;
    if (m < M) {
      int c0 = n0 + wn + nl, c1 = c0 + 16;
      C[m * ldc + c0] = acc0[p] + win_b[c0];
      C[m * ldc + c1] = acc1[p] + win_b[c1];
    }
  }
}

// ---------------- one-time weight conversion / repack ----------------------
__global__ void cvt_bf16_kernel(const float* __restrict__ src,
                                __bf16* __restrict__ dst, long n) {
  long i = (long)blockIdx.x * 256 + threadIdx.x;
  if (i < n) dst[i] = (__bf16)src[i];
}

__global__ void repack_win_kernel(const float* __restrict__ src,  // [o][i][w]
                                  __bf16* __restrict__ dst) {     // [w][o][i]
  int idx = blockIdx.x * 256 + threadIdx.x;
  if (idx >= 128 * 128 * 7) return;
  int w = idx / 16384, r = idx % 16384;
  int o = r >> 7, i = r & 127;
  dst[idx] = (__bf16)src[o * 128 * 7 + i * 7 + w];
}

// ---------------- pointwise passes (HBM-bound, ~10us each at 23.3 TB/s) ----
__global__ void encode_kernel(const float* __restrict__ flux,
                              const float* __restrict__ dt,
                              const float* __restrict__ flux_w,
                              const float* __restrict__ flux_b,
                              __bf16* __restrict__ x, long M) {
  long idx = (long)blockIdx.x * 256 + threadIdx.x;
  if (idx >= M * 128) return;
  long m = idx >> 7; int c = (int)(idx & 127);
  float val;
  if (c < 64) {
    val = flux[m] * flux_w[c] + flux_b[c];
  } else {
    int i = (c - 64) & 31;
    float d = fabsf(dt[m]) + 1e-6f;
    float a = log1pf(d) * expf(-0.28782313662425572f * (float)(2 * i));
    val = (c < 96) ? sinf(a) : cosf(a);
  }
  x[idx] = (__bf16)val;
}

__global__ void ln_act_kernel(const float* __restrict__ X,
                              const float* __restrict__ g,
                              const float* __restrict__ b,
                              __bf16* __restrict__ Y, long ldy,
                              long M, int gelu) {
  const int lane = threadIdx.x & 31, wave = threadIdx.x >> 5;
  const long m = (long)blockIdx.x * 8 + wave;
  if (m >= M) return;
  const float4 xv = *(const float4*)(X + m * 128 + lane * 4);
  float v[4] = {xv.x, xv.y, xv.z, xv.w};
  float mean = wave_sum(v[0] + v[1] + v[2] + v[3]) * (1.f / 128.f);
  float s2 = 0.f;
#pragma unroll
  for (int i = 0; i < 4; ++i) { float d = v[i] - mean; s2 += d * d; }
  float inv = rsqrtf(wave_sum(s2) * (1.f / 128.f) + 1e-5f);
#pragma unroll
  for (int i = 0; i < 4; ++i) {
    int h = lane * 4 + i;
    float x = (v[i] - mean) * inv * g[h] + b[h];
    if (gelu) x = 0.5f * x * (1.f + erff(x * 0.70710678118654752f));
    Y[m * ldy + h] = (__bf16)x;
  }
}

__global__ void glu_ln_kernel(const float* __restrict__ X,  // (M,256)
                              const float* __restrict__ g,
                              const float* __restrict__ b,
                              __bf16* __restrict__ Y, long M) {
  const int lane = threadIdx.x & 31, wave = threadIdx.x >> 5;
  const long m = (long)blockIdx.x * 8 + wave;
  if (m >= M) return;
  const float4 av = *(const float4*)(X + m * 256 + lane * 4);
  const float4 bv = *(const float4*)(X + m * 256 + 128 + lane * 4);
  float a[4] = {av.x, av.y, av.z, av.w};
  float bb[4] = {bv.x, bv.y, bv.z, bv.w};
  float v[4];
#pragma unroll
  for (int i = 0; i < 4; ++i) v[i] = a[i] * (1.f / (1.f + expf(-bb[i])));
  float mean = wave_sum(v[0] + v[1] + v[2] + v[3]) * (1.f / 128.f);
  float s2 = 0.f;
#pragma unroll
  for (int i = 0; i < 4; ++i) { float d = v[i] - mean; s2 += d * d; }
  float inv = rsqrtf(wave_sum(s2) * (1.f / 128.f) + 1e-5f);
#pragma unroll
  for (int i = 0; i < 4; ++i) {
    int h = lane * 4 + i;
    Y[m * 128 + h] = (__bf16)((v[i] - mean) * inv * g[h] + b[h]);
  }
}

// ---------------- persistent GRU scan: 8 blocks x 16 batch rows ------------
#define SCAN_THREADS 512
#define SCAN_B 16
#define HB_S 136   // padded bf16 h-state k-stride
#define HG_S 392   // padded f32 hg-stage n-stride
#define SCAN_LDS (384 * 128 * 2 + SCAN_B * HB_S * 2 + SCAN_B * 128 * 4 + \
                  SCAN_B * HG_S * 4 + 2 * SCAN_B * 128 * 4)

__global__ void gru_scan_kernel(const float* __restrict__ xg,     // (B*T,384) incl i2h bias
                                const float* __restrict__ Wh,     // (384,128)
                                const float* __restrict__ bh,     // (384)
                                const float* __restrict__ lng,    // (3*128)
                                const float* __restrict__ lnb,
                                const float* __restrict__ resid,  // (B*T,128)
                                float* __restrict__ seq_f,        // (B*T,128)
                                __bf16* __restrict__ seq_b,       // (B*T,128)
                                int T) {
  extern __shared__ char smem[];
  __bf16* WhL = (__bf16*)smem;                                   // 98304B, W[n][k]
  __bf16* hB  = (__bf16*)(smem + 384 * 128 * 2);                 // state bf16
  float*  hF  = (float*)(smem + 384 * 128 * 2 + SCAN_B * HB_S * 2);
  float*  hgs = hF + SCAN_B * 128;                               // hg stage f32
  float*  rz  = hgs + SCAN_B * HG_S;                             // r then z
  const int tid = threadIdx.x, lane = tid & 31, wave = tid >> 5;
  const int bbase = blockIdx.x * SCAN_B;

  for (int i = tid; i < 384 * 128; i += SCAN_THREADS) WhL[i] = (__bf16)Wh[i];
  for (int i = tid; i < SCAN_B * 128; i += SCAN_THREADS) hF[i] = 0.f;
  for (int i = tid; i < SCAN_B * HB_S; i += SCAN_THREADS) hB[i] = (__bf16)0.f;
  __syncthreads();

  for (int t = 0; t < T; ++t) {
    // ---- hg = h(16x128) @ Wh^T -> hgs(16x384), 16 waves over 24 n-tiles ----
    asm volatile("s_setprio 1");
    v16bf afr[4];
#pragma unroll
    for (int kc = 0; kc < 4; ++kc) afr[kc] = load_a_frag(hB, HB_S, 0, kc * 32, lane);
    for (int tile = wave; tile < 24; tile += 16) {
      int n0 = tile * 16;
      v8f acc = {};
#pragma unroll
      for (int kc = 0; kc < 4; ++kc) {
        v16bf bfr = load_b_frag(WhL, 128, n0, kc * 32, lane);
        acc = wmma_bf16(afr[kc], bfr, acc);
      }
      int n = n0 + (lane & 15), mo = (lane >> 4) * 8;
#pragma unroll
      for (int p = 0; p < 8; ++p) hgs[(mo + p) * HG_S + n] = acc[p];
    }
    asm volatile("s_setprio 0");
    __syncthreads();

    // ---- gates r,z: wave = batch row, loop over gate 0/1 ----
    {
      const int  b = wave;
      const long mrow = (long)(bbase + b) * T + t;
      if (t + 1 < T)  // hide next step's only global-load latency
        __builtin_prefetch(xg + (mrow + 1) * 384 + lane * 12, 0, 1);
#pragma unroll
      for (int g = 0; g < 2; ++g) {
        const float4 xv = *(const float4*)(xg + mrow * 384 + g * 128 + lane * 4);
        const float xa[4] = {xv.x, xv.y, xv.z, xv.w};
        float pre[4], s1 = 0.f;
#pragma unroll
        for (int i = 0; i < 4; ++i) {
          int h = lane * 4 + i;
          pre[i] = xa[i] + hgs[b * HG_S + g * 128 + h] + bh[g * 128 + h];
          s1 += pre[i];
        }
        float mean = wave_sum(s1) * (1.f / 128.f);
        float s2 = 0.f;
#pragma unroll
        for (int i = 0; i < 4; ++i) { float d = pre[i] - mean; s2 += d * d; }
        float inv = rsqrtf(wave_sum(s2) * (1.f / 128.f) + 1e-5f);
#pragma unroll
        for (int i = 0; i < 4; ++i) {
          int h = lane * 4 + i;
          float a = (pre[i] - mean) * inv * lng[g * 128 + h] + lnb[g * 128 + h];
          rz[g * SCAN_B * 128 + b * 128 + h] = 1.f / (1.f + expf(-a));
        }
      }
    }
    __syncthreads();

    // ---- gate n + state update + sequence write ----
    {
      const int  b = wave;
      const long mrow = (long)(bbase + b) * T + t;
      const float4 xv = *(const float4*)(xg + mrow * 384 + 256 + lane * 4);
      const float xa[4] = {xv.x, xv.y, xv.z, xv.w};
      float pre[4], s1 = 0.f;
#pragma unroll
      for (int i = 0; i < 4; ++i) {
        int h = lane * 4 + i;
        float r = rz[b * 128 + h];
        pre[i] = xa[i] + r * (hgs[b * HG_S + 256 + h] + bh[256 + h]);
        s1 += pre[i];
      }
      float mean = wave_sum(s1) * (1.f / 128.f);
      float s2 = 0.f;
#pragma unroll
      for (int i = 0; i < 4; ++i) { float d = pre[i] - mean; s2 += d * d; }
      float inv = rsqrtf(wave_sum(s2) * (1.f / 128.f) + 1e-5f);
      const float4 rv = *(const float4*)(resid + mrow * 128 + lane * 4);
      const float ra[4] = {rv.x, rv.y, rv.z, rv.w};
#pragma unroll
      for (int i = 0; i < 4; ++i) {
        int h = lane * 4 + i;
        float a  = (pre[i] - mean) * inv * lng[256 + h] + lnb[256 + h];
        float nn = tanhf(a);
        float z  = rz[SCAN_B * 128 + b * 128 + h];
        float hp = hF[b * 128 + h];
        float hn = (1.f - z) * nn + z * hp + ra[i];
        hF[b * 128 + h]     = hn;
        hB[b * HB_S + h]    = (__bf16)hn;
        seq_f[mrow * 128 + h] = hn;
        seq_b[mrow * 128 + h] = (__bf16)hn;
      }
    }
    __syncthreads();
  }
}

// ---------------- final LN + attention score / pooling ---------------------
__global__ void score_kernel(const float* __restrict__ seq,
                             const float* __restrict__ ng, const float* __restrict__ nb,
                             const float* __restrict__ w1, const float* __restrict__ b1,
                             const float* __restrict__ w2, const float* __restrict__ b2,
                             float* __restrict__ seqn, float* __restrict__ s, long M) {
  __shared__ float sh[8 * 128];
  const int lane = threadIdx.x & 31, wave = threadIdx.x >> 5;
  const long m = (long)blockIdx.x * 8 + wave;
  if (m >= M) return;
  float* row = sh + wave * 128;
  const float4 xv = *(const float4*)(seq + m * 128 + lane * 4);
  float v[4] = {xv.x, xv.y, xv.z, xv.w};
  float mean = wave_sum(v[0] + v[1] + v[2] + v[3]) * (1.f / 128.f);
  float s2 = 0.f;
#pragma unroll
  for (int i = 0; i < 4; ++i) { float d = v[i] - mean; s2 += d * d; }
  float inv = rsqrtf(wave_sum(s2) * (1.f / 128.f) + 1e-5f);
#pragma unroll
  for (int i = 0; i < 4; ++i) {
    int h = lane * 4 + i;
    float x = (v[i] - mean) * inv * ng[h] + nb[h];
    row[h] = x;
    seqn[m * 128 + h] = x;
  }
  float sacc = 0.f;
#pragma unroll
  for (int jj = 0; jj < 2; ++jj) {
    int j = lane + jj * 32;
    float d = b1[j];
    for (int h = 0; h < 128; ++h) d += w1[j * 128 + h] * row[h];
    sacc += w2[j] * tanhf(d);
  }
  sacc = wave_sum(sacc);
  if (lane == 0) s[m] = sacc + b2[0];
}

__global__ void pool_kernel(const float* __restrict__ s, const float* __restrict__ mask,
                            const float* __restrict__ seqn, float* __restrict__ out,
                            int T) {
  __shared__ float wbuf[2048];
  __shared__ float red[256];
  const int b = blockIdx.x, tid = threadIdx.x;
  const long base = (long)b * T;
  float mx = -3.0e38f;
  for (int t = tid; t < T; t += 256) {
    float sv = (mask[base + t] > 0.f) ? s[base + t] : -1e9f;
    wbuf[t] = sv;
    mx = fmaxf(mx, sv);
  }
  red[tid] = mx; __syncthreads();
  for (int off = 128; off; off >>= 1) {
    if (tid < off) red[tid] = fmaxf(red[tid], red[tid + off]);
    __syncthreads();
  }
  mx = red[0]; __syncthreads();
  float lsum = 0.f;
  for (int t = tid; t < T; t += 256) { float e = expf(wbuf[t] - mx); wbuf[t] = e; lsum += e; }
  red[tid] = lsum; __syncthreads();
  for (int off = 128; off; off >>= 1) {
    if (tid < off) red[tid] += red[tid + off];
    __syncthreads();
  }
  float Z = red[0]; __syncthreads();
  float l2 = 0.f;
  for (int t = tid; t < T; t += 256) {
    float wv = wbuf[t] / Z * mask[base + t];
    wbuf[t] = wv; l2 += wv;
  }
  red[tid] = l2; __syncthreads();
  for (int off = 128; off; off >>= 1) {
    if (tid < off) red[tid] += red[tid + off];
    __syncthreads();
  }
  float scale = 1.f / (red[0] + 1e-8f);
  __syncthreads();
  for (int t = tid; t < T; t += 256) wbuf[t] *= scale;
  __syncthreads();
  if (tid < 128) {
    float acc = 0.f;
    for (int t = 0; t < T; ++t) acc += wbuf[t] * seqn[(base + t) * 128 + tid];
    out[b * 128 + tid] = acc;
  }
}

// ---------------- host orchestration ---------------------------------------
static inline long cdivl(long a, long b) { return (a + b - 1) / b; }

extern "C" void kernel_launch(void* const* d_in, const int* in_sizes, int n_in,
                              void* d_out, int out_size, void* d_ws, size_t ws_size,
                              hipStream_t stream) {
  (void)n_in; (void)out_size; (void)ws_size;
  const long M = (long)in_sizes[0];   // B*T
  const int  Bn = 128;
  const int  T = (int)(M / Bn);

  const float* flux    = (const float*)d_in[0];
  const float* delta_t = (const float*)d_in[1];
  const float* mask    = (const float*)d_in[2];
  const float* flux_w  = (const float*)d_in[3];
  const float* flux_b  = (const float*)d_in[4];
  const float* mix_w   = (const float*)d_in[5];
  const float* mix_b   = (const float*)d_in[6];
  const float* mix_ln_g = (const float*)d_in[7];
  const float* mix_ln_b = (const float*)d_in[8];
  const float* fe_w1   = (const float*)d_in[9];
  const float* fe_b1   = (const float*)d_in[10];
  const float* fe_ln1_g = (const float*)d_in[11];
  const float* fe_ln1_b = (const float*)d_in[12];
  const float* fe_w2   = (const float*)d_in[13];
  const float* fe_b2   = (const float*)d_in[14];
  const float* fe_ln2_g = (const float*)d_in[15];
  const float* fe_ln2_b = (const float*)d_in[16];
  const float* win_w   = (const float*)d_in[17];
  const float* win_b   = (const float*)d_in[18];
  const float* win_ln_g = (const float*)d_in[19];
  const float* win_ln_b = (const float*)d_in[20];
  const float* i2h_w0  = (const float*)d_in[21];
  const float* i2h_b0  = (const float*)d_in[22];
  const float* h2h_w0  = (const float*)d_in[23];
  const float* h2h_b0  = (const float*)d_in[24];
  const float* ln_g0   = (const float*)d_in[25];
  const float* ln_b0   = (const float*)d_in[26];
  const float* i2h_w   = (const float*)d_in[27];
  const float* i2h_b   = (const float*)d_in[28];
  const float* h2h_w   = (const float*)d_in[29];
  const float* h2h_b   = (const float*)d_in[30];
  const float* ln_g    = (const float*)d_in[31];
  const float* ln_b    = (const float*)d_in[32];
  const float* res_w   = (const float*)d_in[33];
  const float* norm_g  = (const float*)d_in[34];
  const float* norm_b  = (const float*)d_in[35];
  const float* pool_w1 = (const float*)d_in[36];
  const float* pool_b1 = (const float*)d_in[37];
  const float* pool_w2 = (const float*)d_in[38];
  const float* pool_b2 = (const float*)d_in[39];
  float* out = (float*)d_out;

  // workspace arena
  char* ws = (char*)d_ws;
  size_t off = 0;
  auto arena = [&](size_t bytes) {
    void* p = ws + off;
    off += (bytes + 255) & ~(size_t)255;
    return p;
  };
  __bf16* x_bf   = (__bf16*)arena((size_t)M * 128 * 2);  // reused bf16 scratch
  float*  bufF   = (float*)arena((size_t)M * 384 * 4);   // f32 GEMM out / xg
  __bf16* rnn    = (__bf16*)arena((size_t)M * 256 * 2);  // [feat | wout] bf16
  float*  resF   = (float*)arena((size_t)M * 128 * 4);   // residual, later seqn
  float*  seqAf  = (float*)arena((size_t)M * 128 * 4);
  __bf16* seqAb  = (__bf16*)arena((size_t)M * 128 * 2);
  float*  seqBf  = (float*)arena((size_t)M * 128 * 4);
  __bf16* seqBb  = (__bf16*)arena((size_t)M * 128 * 2);
  float*  sbuf   = (float*)arena((size_t)M * 4);
  // bf16 weight cache (converted once per launch)
  __bf16* w_mix  = (__bf16*)arena(16384 * 2);
  __bf16* w_fe1  = (__bf16*)arena(32768 * 2);
  __bf16* w_fe2  = (__bf16*)arena(16384 * 2);
  __bf16* w_win  = (__bf16*)arena(114688 * 2);   // repacked [w][o][i]
  __bf16* w_i2h0 = (__bf16*)arena(98304 * 2);
  __bf16* w_res  = (__bf16*)arena(32768 * 2);
  __bf16* w_i2h  = (__bf16*)arena(98304 * 2);    // both layers

  hipFuncSetAttribute((const void*)gru_scan_kernel,
                      hipFuncAttributeMaxDynamicSharedMemorySize, SCAN_LDS);

  const dim3 blk256(256);
  const long rowsBlocks = cdivl(M, 8);

  // 0) one-time weight conversion to bf16 (tiny: ~0.8 MB total)
  cvt_bf16_kernel<<<dim3(cdivl(16384, 256)), blk256, 0, stream>>>(mix_w, w_mix, 16384);
  cvt_bf16_kernel<<<dim3(cdivl(32768, 256)), blk256, 0, stream>>>(fe_w1, w_fe1, 32768);
  cvt_bf16_kernel<<<dim3(cdivl(16384, 256)), blk256, 0, stream>>>(fe_w2, w_fe2, 16384);
  repack_win_kernel<<<dim3(cdivl(114688, 256)), blk256, 0, stream>>>(win_w, w_win);
  cvt_bf16_kernel<<<dim3(cdivl(98304, 256)), blk256, 0, stream>>>(i2h_w0, w_i2h0, 98304);
  cvt_bf16_kernel<<<dim3(cdivl(32768, 256)), blk256, 0, stream>>>(res_w, w_res, 32768);
  cvt_bf16_kernel<<<dim3(cdivl(98304, 256)), blk256, 0, stream>>>(i2h_w, w_i2h, 98304);

  // 1) encode -> x_bf (M,128)
  encode_kernel<<<dim3(cdivl(M * 128, 256)), blk256, 0, stream>>>(
      flux, delta_t, flux_w, flux_b, x_bf, M);
  // 2) mix GEMM + LN + GELU -> x_bf (through bufF)
  gemm_bf16_kernel<<<dim3(2, cdivl(M, BM)), blk256, 0, stream>>>(
      x_bf, 128, w_mix, 128, mix_b, bufF, 128, M, 128, 128);
  ln_act_kernel<<<dim3(rowsBlocks), blk256, 0, stream>>>(
      bufF, mix_ln_g, mix_ln_b, x_bf, 128, M, 1);
  // 3) fe1 GEMM (N=256) -> GLU + LN -> x_bf
  gemm_bf16_kernel<<<dim3(4, cdivl(M, BM)), blk256, 0, stream>>>(
      x_bf, 128, w_fe1, 128, fe_b1, bufF, 256, M, 256, 128);
  glu_ln_kernel<<<dim3(rowsBlocks), blk256, 0, stream>>>(
      bufF, fe_ln1_g, fe_ln1_b, x_bf, M);
  // 4) fe2 GEMM + LN + GELU -> feat into rnn[:,0:128]
  gemm_bf16_kernel<<<dim3(2, cdivl(M, BM)), blk256, 0, stream>>>(
      x_bf, 128, w_fe2, 128, fe_b2, bufF, 128, M, 128, 128);
  ln_act_kernel<<<dim3(rowsBlocks), blk256, 0, stream>>>(
      bufF, fe_ln2_g, fe_ln2_b, rnn, 256, M, 1);
  // 5) causal window conv + LN + GELU -> wout into rnn[:,128:256]
  conv_gemm_kernel<<<dim3(2, cdivl(M, BM)), blk256, 0, stream>>>(
      rnn, 256, w_win, win_b, bufF, 128, M, T);
  ln_act_kernel<<<dim3(rowsBlocks), blk256, 0, stream>>>(
      bufF, win_ln_g, win_ln_b, rnn + 128, 256, M, 1);
  // 6) xg0 = rnn_in @ i2h_w0^T + b  (K=256, N=384); res = rnn_in @ res_w^T
  gemm_bf16_kernel<<<dim3(6, cdivl(M, BM)), blk256, 0, stream>>>(
      rnn, 256, w_i2h0, 256, i2h_b0, bufF, 384, M, 384, 256);
  gemm_bf16_kernel<<<dim3(2, cdivl(M, BM)), blk256, 0, stream>>>(
      rnn, 256, w_res, 256, nullptr, resF, 128, M, 128, 256);
  // 7) layer-0 scan (8 persistent blocks, batch-split, weights LDS-resident)
  gru_scan_kernel<<<dim3(Bn / SCAN_B), dim3(SCAN_THREADS), SCAN_LDS, stream>>>(
      bufF, h2h_w0, h2h_b0, ln_g0, ln_b0, resF, seqAf, seqAb, T);
  // 8) layer-1 i2h + scan (residual = layer-0 output)
  gemm_bf16_kernel<<<dim3(6, cdivl(M, BM)), blk256, 0, stream>>>(
      seqAb, 128, w_i2h, 128, i2h_b, bufF, 384, M, 384, 128);
  gru_scan_kernel<<<dim3(Bn / SCAN_B), dim3(SCAN_THREADS), SCAN_LDS, stream>>>(
      bufF, h2h_w, h2h_b, ln_g, ln_b, seqAf, seqBf, seqBb, T);
  // 9) layer-2 i2h + scan
  gemm_bf16_kernel<<<dim3(6, cdivl(M, BM)), blk256, 0, stream>>>(
      seqBb, 128, w_i2h + 384 * 128, 128, i2h_b + 384, bufF, 384, M, 384, 128);
  gru_scan_kernel<<<dim3(Bn / SCAN_B), dim3(SCAN_THREADS), SCAN_LDS, stream>>>(
      bufF, h2h_w + 384 * 128, h2h_b + 384, ln_g + 384, ln_b + 384,
      seqBf, seqAf, seqAb, T);
  // 10) final LN + attention scores; seqn reuses resF
  score_kernel<<<dim3(rowsBlocks), blk256, 0, stream>>>(
      seqAf, norm_g, norm_b, pool_w1, pool_b1, pool_w2, pool_b2, resF, sbuf, M);
  // 11) masked softmax pooling -> out (B,128)
  pool_kernel<<<dim3(Bn), blk256, 0, stream>>>(sbuf, mask, resF, out, T);
}
